// MinkUNet_32478542692461
// MI455X (gfx1250) — matrix operand order repaired and checked by reading, not compiled, not run on hardware
//
#include <hip/hip_runtime.h>
#include <hip/hip_bf16.h>

typedef __attribute__((ext_vector_type(16))) _Float16 v16h;
typedef __attribute__((ext_vector_type(8)))  _Float16 v8h;
typedef __attribute__((ext_vector_type(8)))  float    v8f;

#define N_PTS 400000
#define M_VOX 300000
#define KT    27
#define KDEV  8
#define NCLS  19
#define EPSC  1e-5f
#define TROWS 3                 // 16-row tiles per wave (48 rows; 300000 % 48 == 0)

// ---------------------------------------------------------------- utilities
__global__ void k_zero(float* p, int n) {
    int i = blockIdx.x * 256 + threadIdx.x;
    if (i < n) p[i] = 0.f;
}

// Pack W (KT, cinw, 32) f32 -> WMMA B-fragment layout, f16.
// For tap kt, tile t (output cols t*16..t*16+15), lane L:
//   16 halfs, half h -> B[k = (L>>4)*16 + h][n = (L&15) + t*16]
__global__ void k_pack(const float* __restrict__ W, _Float16* __restrict__ wp, int cinw) {
    int t = blockIdx.x * 64 + threadIdx.x;          // KT*2*32 = 1728 threads
    if (t >= KT * 2 * 32) return;
    int lane = t & 31;
    int tile = (t >> 5) & 1;
    int kt   = t >> 6;
    int nn    = (lane & 15) + tile * 16;
    int kbase = (lane >> 4) * 16;
    _Float16* dst = wp + (size_t)t * 16;
    for (int h = 0; h < 16; ++h) {
        int kin = kbase + h;
        float v = (kin < cinw) ? W[(size_t)kt * cinw * 32 + (size_t)kin * 32 + nn] : 0.f;
        dst[h] = (_Float16)v;
    }
}

// ---------------------------------------------------------------- voxelize
__global__ void k_voxelize(const float* __restrict__ pf, const int* __restrict__ idxq,
                           float* __restrict__ vacc, float* __restrict__ cnt) {
    int n = blockIdx.x * 256 + threadIdx.x;
    if (n >= N_PTS) return;
    int q = idxq[n];
    atomicAdd(&cnt[q], 1.f);
#pragma unroll
    for (int c = 0; c < 4; ++c) atomicAdd(&vacc[q * 4 + c], pf[n * 4 + c]);
}

__global__ void k_vox_fin(const float* __restrict__ vacc, const float* __restrict__ cnt,
                          _Float16* __restrict__ x16) {
    int m = blockIdx.x * 256 + threadIdx.x;
    if (m >= M_VOX) return;
    float inv = 1.f / fmaxf(cnt[m], 1.f);
    _Float16* r = x16 + (size_t)m * 32;
#pragma unroll
    for (int c = 0; c < 4; ++c)  r[c] = (_Float16)(vacc[m * 4 + c] * inv);
#pragma unroll
    for (int c = 4; c < 32; ++c) r[c] = (_Float16)0.f;
}

// ---------------------------------------------------------------- sparse conv (WMMA)
// One wave32 per 48-row tile (3x16). B fragments held in registers across the
// 3 row-tiles -> ~2.2 VMEM per WMMA. Fused per-channel BN sum/sumsq.
__global__ void __launch_bounds__(256)
k_conv(const _Float16* __restrict__ x16, const _Float16* __restrict__ wp,
       const int* __restrict__ nbrs, float* __restrict__ pre, float* __restrict__ stats) {
    __shared__ float ls[64];
    int tid = threadIdx.x;
    if (tid < 64) ls[tid] = 0.f;
    __syncthreads();

    int lane = tid & 31;
    int wave = tid >> 5;
    int tb   = (blockIdx.x * 8 + wave) * (16 * TROWS);
    if (tb < M_VOX) {                               // M_VOX % 48 == 0: wave fully valid
        int r0 = tb + (lane & 15);
        int hf = lane >> 4;                         // lane-half selects A channel range
        int cA = hf * 16;
        v8f acc[2 * TROWS] = {};
        v16h vzero = {};
        for (int k = 0; k < KT; ++k) {
            const _Float16* wb = wp + ((size_t)(k * 2) * 32 + lane) * 16;
            v16h b0 = *(const v16h*)wb;             // output cols 0..15
            v16h b1 = *(const v16h*)(wb + 512);     // output cols 16..31
#pragma unroll
            for (int t = 0; t < TROWS; ++t) {
                int nb = nbrs[(r0 + 16 * t) * KT + k];
                int id = nb < 0 ? 0 : nb;
                const _Float16* xr = x16 + (size_t)id * 32;
                v8h lo = *(const v8h*)(xr + cA);        // channels [cA, cA+8)
                v8h hi = *(const v8h*)(xr + cA + 8);    // channels [cA+8, cA+16)
                v16h a;
#pragma unroll
                for (int i = 0; i < 8; ++i) { a[i] = lo[i]; a[i + 8] = hi[i]; }
                if (nb < 0) a = vzero;                  // masked neighbor -> zero row
                acc[2 * t] = __builtin_amdgcn_wmma_f32_16x16x32_f16(
                    false, a, false, b0, (short)0, acc[2 * t], false, false);
                acc[2 * t + 1] = __builtin_amdgcn_wmma_f32_16x16x32_f16(
                    false, a, false, b1, (short)0, acc[2 * t + 1], false, false);
            }
        }
        // epilogue: store 48x32 f32 + per-channel BN partials
        int col = lane & 15;
        float s0 = 0.f, q0 = 0.f, s1 = 0.f, q1 = 0.f;
#pragma unroll
        for (int t = 0; t < TROWS; ++t) {
            int rbase = tb + 16 * t + hf * 8;           // C/D layout: VGPR i -> row hf*8+i
#pragma unroll
            for (int i = 0; i < 8; ++i) {
                float v0 = acc[2 * t][i], v1 = acc[2 * t + 1][i];
                pre[(size_t)(rbase + i) * 32 + col]      = v0;
                pre[(size_t)(rbase + i) * 32 + 16 + col] = v1;
                s0 += v0; q0 += v0 * v0;
                s1 += v1; q1 += v1 * v1;
            }
        }
        atomicAdd(&ls[col],      s0);
        atomicAdd(&ls[32 + col], q0);
        atomicAdd(&ls[16 + col], s1);
        atomicAdd(&ls[48 + col], q1);
    }
    __syncthreads();
    if (tid < 64) atomicAdd(&stats[tid], ls[tid]);
}

// ---------------------------------------------------------------- batch norm
__global__ void k_bnfin(const float* __restrict__ stats, const float* __restrict__ g,
                        const float* __restrict__ b, float* __restrict__ sb) {
    int c = threadIdx.x;
    if (c >= 32) return;
    float mean = stats[c] / (float)M_VOX;
    float var  = stats[32 + c] / (float)M_VOX - mean * mean;
    float sc   = g[c] * rsqrtf(var + EPSC);
    sb[c]      = sc;
    sb[32 + c] = b[c] - mean * sc;
}

__global__ void k_bnrelu(const float* __restrict__ pre, const float* __restrict__ sb,
                         _Float16* __restrict__ y) {
    int i = blockIdx.x * 256 + threadIdx.x;
    if (i >= M_VOX * 32) return;
    int c = i & 31;
    float v = pre[i] * sb[c] + sb[32 + c];
    y[i] = (_Float16)fmaxf(v, 0.f);
}

__global__ void k_bnaddrelu(const float* __restrict__ pre, const float* __restrict__ sb,
                            const _Float16* __restrict__ hres, _Float16* __restrict__ y) {
    int i = blockIdx.x * 256 + threadIdx.x;
    if (i >= M_VOX * 32) return;
    int c = i & 31;
    float v = pre[i] * sb[c] + sb[32 + c] + (float)hres[i];
    y[i] = (_Float16)fmaxf(v, 0.f);
}

// ---------------------------------------------------------------- devox + classifier
__global__ void k_devox(const _Float16* __restrict__ h, const int* __restrict__ idxd,
                        const float* __restrict__ wd, const float* __restrict__ Wc,
                        const float* __restrict__ bc, float* __restrict__ out) {
    int n = blockIdx.x * 256 + threadIdx.x;
    if (n >= N_PTS) return;
    float p[32];
#pragma unroll
    for (int c = 0; c < 32; ++c) p[c] = 0.f;
#pragma unroll
    for (int j = 0; j < KDEV; ++j) {
        int id = idxd[n * KDEV + j];
        if (id >= 0) {
            float w = wd[n * KDEV + j];
            const _Float16* hr = h + (size_t)id * 32;
#pragma unroll
            for (int c = 0; c < 32; ++c) p[c] += w * (float)hr[c];
        }
    }
    for (int c = 0; c < NCLS; ++c) {
        float acc = bc[c];
#pragma unroll
        for (int cc = 0; cc < 32; ++cc) acc += p[cc] * Wc[cc * NCLS + c];
        out[(size_t)n * NCLS + c] = acc;
    }
}

// ---------------------------------------------------------------- launch
extern "C" void kernel_launch(void* const* d_in, const int* in_sizes, int n_in,
                              void* d_out, int out_size, void* d_ws, size_t ws_size,
                              hipStream_t stream) {
    const float* point_fea = (const float*)d_in[0];
    const int*   idx_query = (const int*)d_in[1];
    const int*   nbrs      = (const int*)d_in[2];
    const int*   idx_dev   = (const int*)d_in[3];
    const float* w_dev     = (const float*)d_in[4];
    const float* W_s1 = (const float*)d_in[5];
    const float* W_s2 = (const float*)d_in[6];
    const float* g_s1 = (const float*)d_in[7];
    const float* b_s1 = (const float*)d_in[8];
    const float* g_s2 = (const float*)d_in[9];
    const float* b_s2 = (const float*)d_in[10];
    const float* W_r1 = (const float*)d_in[11];
    const float* W_r2 = (const float*)d_in[12];
    const float* g_r1 = (const float*)d_in[13];
    const float* b_r1 = (const float*)d_in[14];
    const float* g_r2 = (const float*)d_in[15];
    const float* b_r2 = (const float*)d_in[16];
    const float* W_c  = (const float*)d_in[17];
    const float* b_c  = (const float*)d_in[18];
    float* out = (float*)d_out;

    // ---- workspace layout (~96.3 MB)
    char* ws = (char*)d_ws;
    float* pre  = (float*)ws;                              // M*32 f32  (38.4 MB)
    float* vacc = (float*)ws;                              // aliased: M*4 f32 (pre-conv phase)
    float* cnt  = (float*)(ws + (size_t)M_VOX * 4 * 4);    // aliased: M f32
    size_t off = (size_t)M_VOX * 32 * 4;
    _Float16* bufA = (_Float16*)(ws + off); off += (size_t)M_VOX * 32 * 2;  // vox / r1
    _Float16* bufB = (_Float16*)(ws + off); off += (size_t)M_VOX * 32 * 2;  // h1 / hF
    _Float16* bufC = (_Float16*)(ws + off); off += (size_t)M_VOX * 32 * 2;  // h2 (residual)
    const size_t WPSZ = (size_t)KT * 2 * 32 * 16;          // halfs per packed weight set
    _Float16* wp1 = (_Float16*)(ws + off); off += WPSZ * 2;
    _Float16* wp2 = (_Float16*)(ws + off); off += WPSZ * 2;
    _Float16* wp3 = (_Float16*)(ws + off); off += WPSZ * 2;
    _Float16* wp4 = (_Float16*)(ws + off); off += WPSZ * 2;
    float* stats = (float*)(ws + off); off += 4 * 64 * 4;  // sum/sumsq per conv
    float* sb    = (float*)(ws + off); off += 4 * 64 * 4;  // scale/shift per conv

    const int ROWS_PER_BLK = 8 * 16 * TROWS;               // 384
    const int CONV_GRID = (M_VOX + ROWS_PER_BLK - 1) / ROWS_PER_BLK;  // 782
    const int EW_GRID   = (M_VOX * 32 + 255) / 256;        // 37500
    const int PT_GRID   = (N_PTS + 255) / 256;             // 1563

    // zero accumulators (atomics accumulate -> must re-zero every call)
    k_zero<<<(M_VOX * 5 + 255) / 256, 256, 0, stream>>>((float*)ws, M_VOX * 5);
    k_zero<<<1, 256, 0, stream>>>(stats, 256);

    // pack weights into WMMA B-fragment layout
    k_pack<<<27, 64, 0, stream>>>(W_s1, wp1, 4);
    k_pack<<<27, 64, 0, stream>>>(W_s2, wp2, 32);
    k_pack<<<27, 64, 0, stream>>>(W_r1, wp3, 32);
    k_pack<<<27, 64, 0, stream>>>(W_r2, wp4, 32);

    // voxelize (scatter-mean), zero-padded to 32 channels
    k_voxelize<<<PT_GRID, 256, 0, stream>>>(point_fea, idx_query, vacc, cnt);
    k_vox_fin<<<(M_VOX + 255) / 256, 256, 0, stream>>>(vacc, cnt, bufA);

    // conv1: vox -> h1
    k_conv<<<CONV_GRID, 256, 0, stream>>>(bufA, wp1, nbrs, pre, stats + 0);
    k_bnfin<<<1, 32, 0, stream>>>(stats + 0, g_s1, b_s1, sb + 0);
    k_bnrelu<<<EW_GRID, 256, 0, stream>>>(pre, sb + 0, bufB);

    // conv2: h1 -> h2 (kept for residual)
    k_conv<<<CONV_GRID, 256, 0, stream>>>(bufB, wp2, nbrs, pre, stats + 64);
    k_bnfin<<<1, 32, 0, stream>>>(stats + 64, g_s2, b_s2, sb + 64);
    k_bnrelu<<<EW_GRID, 256, 0, stream>>>(pre, sb + 64, bufC);

    // conv3: h2 -> r1
    k_conv<<<CONV_GRID, 256, 0, stream>>>(bufC, wp3, nbrs, pre, stats + 128);
    k_bnfin<<<1, 32, 0, stream>>>(stats + 128, g_r1, b_r1, sb + 128);
    k_bnrelu<<<EW_GRID, 256, 0, stream>>>(pre, sb + 128, bufA);

    // conv4: r1 -> bn -> +h2 -> relu -> hF
    k_conv<<<CONV_GRID, 256, 0, stream>>>(bufA, wp4, nbrs, pre, stats + 192);
    k_bnfin<<<1, 32, 0, stream>>>(stats + 192, g_r2, b_r2, sb + 192);
    k_bnaddrelu<<<EW_GRID, 256, 0, stream>>>(pre, sb + 192, bufC, bufB);

    // devoxelize + classifier
    k_devox<<<PT_GRID, 256, 0, stream>>>(bufB, idx_dev, w_dev, W_c, b_c, out);
}
